// DGANet_25469156065428
// MI455X (gfx1250) — matrix operand order repaired and checked by reading, not compiled, not run on hardware
//
#include <hip/hip_runtime.h>
#include <hip/hip_bf16.h>
#include <math.h>

// ---------------------------------------------------------------------------
// DGANet dual-GAT forward for MI455X (gfx1250, wave32, WMMA).
//   branch(b): Wh = h @ W_b            (WMMA f16->f32)
//              s1 = Wh@a1, s2 = Wh@a2  (wave reductions)
//              stats: row max / sum of exp(leakyrelu(s1_i+s2_j)) over adj_b
//              out_b = elu( softmax_tile-on-the-fly @ Wh )   (WMMA, LDS tiles)
//   out = out_n + out_d
// ---------------------------------------------------------------------------

typedef __attribute__((ext_vector_type(16))) _Float16 v16h;
typedef __attribute__((ext_vector_type(8)))  float    v8f;

#define NN    4096
#define FIN   512
#define FOUT  256
#define ALPHA 0.2f

static __device__ __forceinline__ float lrelu(float x) { return x > 0.f ? x : ALPHA * x; }

// ===========================================================================
// Kernel A: Wh = h @ W for both branches.
// One wave per 16x16 output tile, K loop of 16 steps (K=32 per WMMA).
// Stores Wh fp32 [b][row][col] and WhT f16 [b][col][row] (for attn B-operand).
// ===========================================================================
__global__ void k_gemm_wh(const float* __restrict__ h,
                          const float* __restrict__ W_n,
                          const float* __restrict__ W_d,
                          float* __restrict__ Wh,
                          _Float16* __restrict__ WhT)
{
    const int wave  = threadIdx.x >> 5;
    const int lane  = threadIdx.x & 31;
    const int nTile = blockIdx.x * 4 + wave;   // 0..15
    const int i0    = blockIdx.y * 16;
    const int b     = blockIdx.z;
    const float* __restrict__ W = b ? W_d : W_n;

    const int m    = lane & 15;   // A row / B col / D col index
    const int half = lane >> 4;

    v8f acc = {};
    for (int kt = 0; kt < FIN / 32; ++kt) {
        const int k0 = kt * 32;
        v16h a, bm;
        // A operand: 16-bit A 16x32 layout — lane half selects K sub-range.
        const float* hrow = h + (size_t)(i0 + m) * FIN + k0;
        #pragma unroll
        for (int v = 0; v < 8; ++v) {
            int kk = ((v & 4) ? 16 : 0) + half * 8 + (v & 3) * 2;
            float2 hv = *(const float2*)(hrow + kk);
            a[2 * v]     = (_Float16)hv.x;
            a[2 * v + 1] = (_Float16)hv.y;
        }
        // B operand: 16-bit B 32x16 — same K order as A (lane half selects K sub-range).
        const float* wcol = W + (size_t)k0 * FOUT + nTile * 16 + m;
        #pragma unroll
        for (int v = 0; v < 8; ++v) {
            int kk = ((v & 4) ? 16 : 0) + half * 8 + (v & 3) * 2;
            bm[2 * v]     = (_Float16)wcol[(size_t)kk * FOUT];
            bm[2 * v + 1] = (_Float16)wcol[(size_t)(kk + 1) * FOUT];
        }
        acc = __builtin_amdgcn_wmma_f32_16x16x32_f16(false, a, false, bm,
                                                     (short)0, acc, false, false);
    }

    // D layout: lane l, VGPR v -> (row = v + 8*(l>>4), col = l&15)
    const int col = nTile * 16 + m;
    union { _Float16 hs[8]; uint4 u; } pk;
    #pragma unroll
    for (int v = 0; v < 8; ++v) {
        int row = i0 + half * 8 + v;
        Wh[(size_t)b * NN * FOUT + (size_t)row * FOUT + col] = acc[v];
        pk.hs[v] = (_Float16)acc[v];
    }
    // rows contiguous across v -> single 16B store into transposed f16 copy
    *(uint4*)(WhT + (size_t)b * FOUT * NN + (size_t)col * NN + i0 + half * 8) = pk.u;
}

// ===========================================================================
// Kernel B: s1[r] = Wh[r,:]@a1, s2[r] = Wh[r,:]@a2. One wave per row.
// ===========================================================================
__global__ void k_proj_s(const float* __restrict__ Wh,
                         const float* __restrict__ a1_n, const float* __restrict__ a2_n,
                         const float* __restrict__ a1_d, const float* __restrict__ a2_d,
                         float* __restrict__ s1, float* __restrict__ s2)
{
    const int wave = threadIdx.x >> 5;
    const int lane = threadIdx.x & 31;
    const int row  = blockIdx.x * 8 + wave;
    const int b    = blockIdx.z;
    const float* __restrict__ a1 = b ? a1_d : a1_n;
    const float* __restrict__ a2 = b ? a2_d : a2_n;
    const float* __restrict__ wr = Wh + (size_t)b * NN * FOUT + (size_t)row * FOUT;

    float acc1 = 0.f, acc2 = 0.f;
    #pragma unroll
    for (int t = 0; t < FOUT / 32; ++t) {
        int n = lane + t * 32;
        float x = wr[n];
        acc1 += x * a1[n];
        acc2 += x * a2[n];
    }
    #pragma unroll
    for (int off = 16; off > 0; off >>= 1) {
        acc1 += __shfl_xor(acc1, off, 32);
        acc2 += __shfl_xor(acc2, off, 32);
    }
    if (lane == 0) { s1[b * NN + row] = acc1; s2[b * NN + row] = acc2; }
}

// ===========================================================================
// Kernel C: masked-softmax stats per row (online max/sum; one wave per row).
// leakyrelu is monotone, so a straight online scan is exact.
// ===========================================================================
__global__ void k_softmax_stats(const int* __restrict__ adj_n,
                                const int* __restrict__ adj_d,
                                const float* __restrict__ s1,
                                const float* __restrict__ s2,
                                float* __restrict__ rmax,
                                float* __restrict__ rsum)
{
    const int wave = threadIdx.x >> 5;
    const int lane = threadIdx.x & 31;
    const int row  = blockIdx.x * 8 + wave;
    const int b    = blockIdx.z;
    const int* __restrict__ adj = (b ? adj_d : adj_n) + (size_t)row * NN;
    const float* __restrict__ s2b = s2 + b * NN;
    const float c1 = s1[b * NN + row];

    float m = -3.0e38f, sum = 0.f;
    for (int j = lane; j < NN; j += 32) {
        if (adj[j] > 0) {
            float v  = lrelu(c1 + s2b[j]);
            float nm = fmaxf(m, v);
            sum = sum * __expf(m - nm) + __expf(v - nm);
            m = nm;
        }
    }
    #pragma unroll
    for (int off = 16; off > 0; off >>= 1) {
        float om = __shfl_xor(m, off, 32);
        float os = __shfl_xor(sum, off, 32);
        float nm = fmaxf(m, om);
        sum = sum * __expf(m - nm) + os * __expf(om - nm);
        m = nm;
    }
    if (lane == 0) {
        rmax[b * NN + row] = m;
        rsum[b * NN + row] = (sum > 0.f) ? sum : 1.0f;  // empty row guard
    }
}

// ===========================================================================
// Kernel D: R_b = elu( softmax(att) @ Wh ).
// Block = 256 thr (8 waves) handles a 16-row tile across full FOUT=256:
//   each K-step (32 cols of adj) builds the 16x32 f16 probability tile ONCE
//   in LDS; every wave ds_loads it as the WMMA A-operand and pairs it with
//   32 contiguous bytes of WhT as the B-operand. adj read exactly once here.
// ===========================================================================
__global__ void k_attn_out(const int* __restrict__ adj_n,
                           const int* __restrict__ adj_d,
                           const float* __restrict__ s1,
                           const float* __restrict__ s2,
                           const float* __restrict__ rmax,
                           const float* __restrict__ rsum,
                           const _Float16* __restrict__ WhT,
                           float* __restrict__ R)
{
    __shared__ __align__(16) _Float16 attT[16 * 32];
    __shared__ float sc1[16], smx[16], sinv[16];

    const int tid  = threadIdx.x;
    const int wave = tid >> 5;
    const int lane = tid & 31;
    const int i0   = blockIdx.y * 16;
    const int b    = blockIdx.z;
    const int* __restrict__ adj   = b ? adj_d : adj_n;
    const float* __restrict__ s2b = s2 + b * NN;

    if (tid < 16) {
        int r = i0 + tid;
        sc1[tid]  = s1[b * NN + r];
        smx[tid]  = rmax[b * NN + r];
        sinv[tid] = 1.0f / rsum[b * NN + r];
    }
    __syncthreads();

    const int m    = lane & 15;
    const int half = lane >> 4;
    const int n0a  = wave * 32;        // this wave's two 16-wide N tiles
    const int n0b  = wave * 32 + 16;
    const _Float16* __restrict__ Bbase = WhT + (size_t)b * FOUT * NN;

    v8f acc0 = {}, acc1v = {};
    for (int kt = 0; kt < NN / 32; ++kt) {
        const int j0 = kt * 32;
        // ---- build 16x32 probability tile cooperatively (2 elems/thread) ----
        for (int e = tid; e < 512; e += 256) {
            int mm = e >> 5, kk = e & 31;
            int j  = j0 + kk;
            float p = 0.f;
            if (adj[(size_t)(i0 + mm) * NN + j] > 0) {
                float v = lrelu(sc1[mm] + s2b[j]);
                p = __expf(v - smx[mm]) * sinv[mm];
            }
            attT[e] = (_Float16)p;
            if (kt + 1 < NN / 32)      // prefetch next adj tile -> global_prefetch
                __builtin_prefetch(adj + (size_t)(i0 + mm) * NN + j + 32, 0, 3);
        }
        __syncthreads();

        // ---- A operand from LDS (two ds_load_b128 per lane) ----
        v16h a;
        {
            const uint4* src = (const uint4*)(attT + m * 32 + half * 16);
            ((uint4*)&a)[0] = src[0];   // K 0..7  (+half*16)
            ((uint4*)&a)[1] = src[1];   // K 8..15 (+half*16)
        }
        // ---- B operands: 32 contiguous bytes of WhT per lane ----
        const _Float16* bp0 = Bbase + (size_t)(n0a + m) * NN + j0 + half * 16;
        const _Float16* bp1 = Bbase + (size_t)(n0b + m) * NN + j0 + half * 16;
        v16h b0, b1;
        ((uint4*)&b0)[0] = ((const uint4*)bp0)[0];
        ((uint4*)&b0)[1] = ((const uint4*)bp0)[1];
        ((uint4*)&b1)[0] = ((const uint4*)bp1)[0];
        ((uint4*)&b1)[1] = ((const uint4*)bp1)[1];

        acc0  = __builtin_amdgcn_wmma_f32_16x16x32_f16(false, a, false, b0,
                                                       (short)0, acc0, false, false);
        acc1v = __builtin_amdgcn_wmma_f32_16x16x32_f16(false, a, false, b1,
                                                       (short)0, acc1v, false, false);
        __syncthreads();
    }

    #pragma unroll
    for (int v = 0; v < 8; ++v) {
        int row  = i0 + half * 8 + v;
        float x0 = acc0[v];
        float x1 = acc1v[v];
        x0 = x0 > 0.f ? x0 : (__expf(x0) - 1.0f);   // ELU
        x1 = x1 > 0.f ? x1 : (__expf(x1) - 1.0f);
        R[(size_t)b * NN * FOUT + (size_t)row * FOUT + n0a + m] = x0;
        R[(size_t)b * NN * FOUT + (size_t)row * FOUT + n0b + m] = x1;
    }
}

// ===========================================================================
// Kernel E: out = R_n + R_d
// ===========================================================================
__global__ void k_combine(const float* __restrict__ R, float* __restrict__ out)
{
    int idx = blockIdx.x * blockDim.x + threadIdx.x;
    out[idx] = R[idx] + R[(size_t)NN * FOUT + idx];
}

// ===========================================================================
extern "C" void kernel_launch(void* const* d_in, const int* in_sizes, int n_in,
                              void* d_out, int out_size, void* d_ws, size_t ws_size,
                              hipStream_t stream)
{
    const float* h     = (const float*)d_in[0];
    const int*   adj_n = (const int*)d_in[1];
    const int*   adj_d = (const int*)d_in[2];
    // d_in[3] = G_nd : computed-but-unused branch in the reference -> skipped
    const float* W_n   = (const float*)d_in[4];
    const float* a1_n  = (const float*)d_in[5];
    const float* a2_n  = (const float*)d_in[6];
    const float* W_d   = (const float*)d_in[7];
    const float* a1_d  = (const float*)d_in[8];
    const float* a2_d  = (const float*)d_in[9];

    // workspace carve-up (~20.25 MB)
    float*    Wh   = (float*)d_ws;                       // 2*NN*FOUT f32
    _Float16* WhT  = (_Float16*)(Wh + 2 * NN * FOUT);    // 2*FOUT*NN f16
    float*    s1   = (float*)(WhT + 2 * (size_t)FOUT * NN);
    float*    s2   = s1 + 2 * NN;
    float*    rmax = s2 + 2 * NN;
    float*    rsum = rmax + 2 * NN;
    float*    R    = rsum + 2 * NN;                      // 2*NN*FOUT f32

    k_gemm_wh     <<<dim3(4, NN / 16, 2), 128, 0, stream>>>(h, W_n, W_d, Wh, WhT);
    k_proj_s      <<<dim3(NN / 8, 1, 2), 256, 0, stream>>>(Wh, a1_n, a2_n, a1_d, a2_d, s1, s2);
    k_softmax_stats<<<dim3(NN / 8, 1, 2), 256, 0, stream>>>(adj_n, adj_d, s1, s2, rmax, rsum);
    k_attn_out    <<<dim3(1, NN / 16, 2), 256, 0, stream>>>(adj_n, adj_d, s1, s2, rmax, rsum, WhT, R);
    k_combine     <<<dim3(NN * FOUT / 256), 256, 0, stream>>>(R, (float*)d_out);
}
